// W8A16Linear_22419729285599
// MI455X (gfx1250) — matrix operand order repaired and checked by reading, not compiled, not run on hardware
//
#include <hip/hip_runtime.h>
#include <stdint.h>

typedef __attribute__((ext_vector_type(16))) _Float16 v16h;
typedef __attribute__((ext_vector_type(8)))  _Float16 v8h;
typedef __attribute__((ext_vector_type(4)))  _Float16 v4h;
typedef __attribute__((ext_vector_type(2)))  _Float16 h2;
typedef __attribute__((ext_vector_type(8)))  float    v8f;
typedef __attribute__((ext_vector_type(4)))  float    v4f;
typedef __attribute__((ext_vector_type(4)))  unsigned v4u;

#define IN_F   4096
#define OUT_F  32000
#define BATCH  16
#define WPB    8      // waves per block (256 threads)

// ---------------------------------------------------------------------------
// Kernel 1: x fp32 -> fp16 once into workspace (64K elements, noise).
// ---------------------------------------------------------------------------
__global__ __launch_bounds__(256) void cvt_x_to_f16(const float* __restrict__ x,
                                                    _Float16* __restrict__ xh) {
    int i = blockIdx.x * blockDim.x + threadIdx.x;     // 16384 threads, 4 elems each
    v4f v = ((const v4f*)x)[i];
    v4h h = { (_Float16)v.x, (_Float16)v.y, (_Float16)v.z, (_Float16)v.w };
    ((v4h*)xh)[i] = h;
}

// int8 dword -> two f16 pairs, fully scalar SSA (no vector subscripts).
// f16(0x6400 | (b ^ 0x80)) == 1152 + b_signed (exact); then -1152.
__device__ __forceinline__ void cvt8(unsigned w, const h2 c, h2& lo, h2& hi) {
    unsigned u = w ^ 0x80808080u;
    lo = __builtin_bit_cast(h2, __builtin_amdgcn_perm(u, 0x64646464u, 0x01050004u)) + c;
    hi = __builtin_bit_cast(h2, __builtin_amdgcn_perm(u, 0x64646464u, 0x00070006u)) + c;
}

// ---------------------------------------------------------------------------
// Kernel 2: one wave = one 16(M) x 16(N) f32 tile over a compile-time K-slice,
// accumulated with v_wmma_f32_16x16x32_f16.
// ---------------------------------------------------------------------------
template <int K_LEN>
__global__ __launch_bounds__(256) void w8a16_wmma_kernel(
    const _Float16* __restrict__ xh,      // [16][4096] f16
    const int8_t*   __restrict__ W,       // [32000][4096] int8
    float*          __restrict__ partial) // [nsplit][16][32000] f32
{
    const int lane = threadIdx.x & 31;
    const int wave = threadIdx.x >> 5;
    const int half = lane >> 4;        // 0: lanes 0-15, 1: lanes 16-31
    const int l16  = lane & 15;

    const int n_tile = blockIdx.x * WPB + wave;
    const int nb     = n_tile * 16;
    const int k0     = blockIdx.y * K_LEN;

    // A: lane holds M=l16; 16 consecutive K starting at half*16
    const _Float16* xrow = xh + (size_t)l16 * IN_F + k0 + half * 16;
    // B: lane holds N=l16; 16 consecutive K starting at half*16
    const int8_t*   wrow = W + (size_t)(nb + l16) * IN_F + k0 + half * 16;

    const h2 c1152 = { (_Float16)-1152.0f, (_Float16)-1152.0f };

    v8f acc = {};

#pragma unroll 8
    for (int kb = 0; kb < K_LEN; kb += 32) {
        // ---- A fragment: two 16B f16 loads (global_load_b128 each) ----
        v8h a0 = *(const v8h*)(xrow + kb);
        v8h a1 = *(const v8h*)(xrow + kb + 8);
        v16h a = __builtin_shufflevector(a0, a1,
                                         0,1,2,3,4,5,6,7, 8,9,10,11,12,13,14,15);

        // ---- B fragment: ONE plain 16B vector load (global_load_b128) ----
        v4u wb = *(const v4u*)(wrow + kb);

        h2 p0, p1, p2, p3, p4, p5, p6, p7;
        cvt8(wb.x, c1152, p0, p1);
        cvt8(wb.y, c1152, p2, p3);
        cvt8(wb.z, c1152, p4, p5);
        cvt8(wb.w, c1152, p6, p7);

        v16h b = { p0.x, p0.y, p1.x, p1.y, p2.x, p2.y, p3.x, p3.y,
                   p4.x, p4.y, p5.x, p5.y, p6.x, p6.y, p7.x, p7.y };

        acc = __builtin_amdgcn_wmma_f32_16x16x32_f16(
                  false, a, false, b, (short)0, acc, false, false);
    }

    // ---- write f32 partial tile (scale/bias fused in reduce kernel) ----
    float* pp = partial + (size_t)blockIdx.y * (BATCH * OUT_F);
    const int n = nb + l16;
#pragma unroll
    for (int rI = 0; rI < 8; ++rI) {
        pp[(size_t)(rI + half * 8) * OUT_F + n] = acc[rI];
    }
}

// ---------------------------------------------------------------------------
// Kernel 3: out[m][n] = (sum_s partial[s][m][n]) * scale[n] + bias[n]
// ---------------------------------------------------------------------------
__global__ __launch_bounds__(256) void reduce_scale_bias(
    const float* __restrict__ partial,
    const float* __restrict__ scale,
    const float* __restrict__ bias,
    float*       __restrict__ out,
    int nsplit)
{
    const int n = blockIdx.x * blockDim.x + threadIdx.x;  // 125*256 = 32000
    const int m = blockIdx.y;                             // 16
    float s = 0.0f;
    for (int sp = 0; sp < nsplit; ++sp)
        s += partial[(size_t)sp * (BATCH * OUT_F) + (size_t)m * OUT_F + n];
    out[(size_t)m * OUT_F + n] = s * scale[n] + bias[n];
}

// ---------------------------------------------------------------------------
extern "C" void kernel_launch(void* const* d_in, const int* in_sizes, int n_in,
                              void* d_out, int out_size, void* d_ws, size_t ws_size,
                              hipStream_t stream) {
    const float*  x     = (const float*)d_in[0];     // [16,1,4096] f32
    const int8_t* W     = (const int8_t*)d_in[1];    // [32000,4096] int8
    const float*  scale = (const float*)d_in[2];     // [32000,1] f32
    const float*  bias  = (const float*)d_in[3];     // [1,32000] f32
    float*        out   = (float*)d_out;             // [16,1,32000] f32

    const size_t part_elems = (size_t)BATCH * OUT_F;          // 512000 f32
    const size_t xh_bytes   = (size_t)BATCH * IN_F * 2;       // 128 KB

    // K-split 2 doubles wave count (latency hiding) for ~8MB extra traffic.
    const int split = (ws_size >= 2 * part_elems * 4 + xh_bytes) ? 2 : 1;

    float*    partial = (float*)d_ws;
    _Float16* xh      = (_Float16*)((char*)d_ws + (size_t)split * part_elems * 4);

    // 1) x fp32 -> fp16
    cvt_x_to_f16<<<(BATCH * IN_F / 4) / 256, 256, 0, stream>>>(x, xh);

    // 2) GEMM: 2000 N-tiles of 16 x split K-slices; 8 waves per block
    if (split == 2) {
        dim3 g(OUT_F / 16 / WPB, 2);
        w8a16_wmma_kernel<IN_F / 2><<<g, 256, 0, stream>>>(xh, W, partial);
    } else {
        dim3 g(OUT_F / 16 / WPB, 1);
        w8a16_wmma_kernel<IN_F><<<g, 256, 0, stream>>>(xh, W, partial);
    }

    // 3) reduce + scale + bias
    dim3 rgrid(OUT_F / 256, BATCH);
    reduce_scale_bias<<<rgrid, 256, 0, stream>>>(partial, scale, bias, out, split);
}
